// Multiheaded_Self_Attention_56410100466002
// MI455X (gfx1250) — matrix operand order repaired and checked by reading, not compiled, not run on hardware
//
#include <hip/hip_runtime.h>

// ---------------------------------------------------------------------------
// MI455X (gfx1250) multi-head self-attention with RoPE, bf16 WMMA pipeline.
// B=4, S=2048, D=1024, H=16, dk=64.
// ---------------------------------------------------------------------------

#define S_LEN 2048
#define D_MODEL 1024
#define N_HEADS 16
#define D_K 64
#define BATCH 4
#define M_ROWS (BATCH * S_LEN)   // 8192

typedef __bf16 v16bf __attribute__((ext_vector_type(16)));
typedef __bf16 v8bf  __attribute__((ext_vector_type(8)));
typedef float  v8f   __attribute__((ext_vector_type(8)));
typedef float  v4f   __attribute__((ext_vector_type(4)));

union ABf { v16bf v; v8bf h[2]; };

__device__ __forceinline__ __bf16 f2bf(float f) {
  unsigned u = __builtin_bit_cast(unsigned, f);
  u = (u + 0x7FFFu + ((u >> 16) & 1u)) >> 16;   // round-to-nearest-even
  return __builtin_bit_cast(__bf16, (unsigned short)u);
}
__device__ __forceinline__ float bf2f(__bf16 b) {
  unsigned u = ((unsigned)__builtin_bit_cast(unsigned short, b)) << 16;
  return __builtin_bit_cast(float, u);
}

#define WMMA_BF16(A, B, C) \
  __builtin_amdgcn_wmma_f32_16x16x32_bf16(false, (A), false, (B), (short)0, (C), false, false)

// 16-lane-group reductions (wave32: masks 1,2,4,8 stay within each half-wave)
__device__ __forceinline__ float rowmax16(float v) {
  v = fmaxf(v, __shfl_xor(v, 1, 32));
  v = fmaxf(v, __shfl_xor(v, 2, 32));
  v = fmaxf(v, __shfl_xor(v, 4, 32));
  v = fmaxf(v, __shfl_xor(v, 8, 32));
  return v;
}
__device__ __forceinline__ float rowsum16(float v) {
  v += __shfl_xor(v, 1, 32);
  v += __shfl_xor(v, 2, 32);
  v += __shfl_xor(v, 4, 32);
  v += __shfl_xor(v, 8, 32);
  return v;
}

// ---------------------------------------------------------------------------
// f32 -> bf16 conversion, 8 elements per thread (2x float4 in, 16B out)
// ---------------------------------------------------------------------------
__global__ __launch_bounds__(256) void cvt_bf16(const float* __restrict__ src,
                                                __bf16* __restrict__ dst) {
  const size_t i = ((size_t)blockIdx.x * blockDim.x + threadIdx.x) * 8;
  v4f a = *(const v4f*)(src + i);
  v4f b = *(const v4f*)(src + i + 4);
  v8bf o;
  o[0] = f2bf(a[0]); o[1] = f2bf(a[1]); o[2] = f2bf(a[2]); o[3] = f2bf(a[3]);
  o[4] = f2bf(b[0]); o[5] = f2bf(b[1]); o[6] = f2bf(b[2]); o[7] = f2bf(b[3]);
  *(v8bf*)(dst + i) = o;
}

// ---------------------------------------------------------------------------
// GEMM: Y = A @ W^T, A:[M,1024] bf16 row-major, W:[1024,1024] bf16 row-major.
// Block = 128 threads (4 waves, 2x2), block tile 128x128, wave tile 64x64
// (4x4 WMMA tiles -> 16 b128 loads per 16 WMMAs per k-step).
// MODE 0: write bf16 to [B,H,S,dk]   (Q, K)
// MODE 1: write bf16 to [B,H,dk,S]   (V transposed)
// MODE 2: write f32  to [M,1024]     (final output projection)
// ---------------------------------------------------------------------------
template <int MODE>
__global__ __launch_bounds__(128) void gemm_bf16(const __bf16* __restrict__ A,
                                                 const __bf16* __restrict__ W,
                                                 void* __restrict__ Y) {
  const int lane = threadIdx.x & 31;
  const int w = threadIdx.x >> 5;
  const int l15 = lane & 15;
  const int hi = (lane >> 4) & 1;
  const int m0 = blockIdx.x * 128 + (w & 1) * 64;
  const int n0 = blockIdx.y * 128 + (w >> 1) * 64;

  v8f acc[4][4] = {};
  const __bf16* arow[4];
  const __bf16* brow[4];
#pragma unroll
  for (int i = 0; i < 4; ++i) {
    arow[i] = A + (size_t)(m0 + i * 16 + l15) * D_MODEL;
    brow[i] = W + (size_t)(n0 + i * 16 + l15) * D_MODEL;
  }
  const int kaOff = hi ? 16 : 0;   // A frag: lanes 16-31 hold K=16..31
  const int kbOff = hi ? 16 : 0;   // B frag: lanes 16-31 hold K=16..31

  for (int k0 = 0; k0 < D_MODEL; k0 += 32) {
    ABf a[4], b[4];
#pragma unroll
    for (int i = 0; i < 4; ++i) {
      a[i].h[0] = *(const v8bf*)(arow[i] + k0 + kaOff);
      a[i].h[1] = *(const v8bf*)(arow[i] + k0 + kaOff + 8);
      b[i].h[0] = *(const v8bf*)(brow[i] + k0 + kbOff);
      b[i].h[1] = *(const v8bf*)(brow[i] + k0 + kbOff + 8);
    }
#pragma unroll
    for (int mi = 0; mi < 4; ++mi)
#pragma unroll
      for (int nj = 0; nj < 4; ++nj)
        acc[mi][nj] = WMMA_BF16(a[mi].v, b[nj].v, acc[mi][nj]);
  }

#pragma unroll
  for (int mi = 0; mi < 4; ++mi)
#pragma unroll
    for (int nj = 0; nj < 4; ++nj)
#pragma unroll
      for (int r = 0; r < 8; ++r) {
        const int m = m0 + mi * 16 + r + hi * 8;  // D layout: M = r + 8*(lane>>4)
        const int n = n0 + nj * 16 + l15;         //           N = lane & 15
        const float v = acc[mi][nj][r];
        if (MODE == 2) {
          ((float*)Y)[(size_t)m * D_MODEL + n] = v;
        } else {
          const int b = m >> 11, s = m & (S_LEN - 1);
          const int h = n >> 6, d = n & (D_K - 1);
          if (MODE == 0)  // [B,H,S,dk]
            ((__bf16*)Y)[(((size_t)(b * N_HEADS + h) * S_LEN) + s) * D_K + d] = f2bf(v);
          else            // [B,H,dk,S]
            ((__bf16*)Y)[((size_t)(b * N_HEADS + h) * D_K + d) * S_LEN + s] = f2bf(v);
        }
      }
}

// ---------------------------------------------------------------------------
// RoPE in-place on Q and K, layout [B,H,S,dk], interleaved pairs (2i, 2i+1).
// One thread per pair; id bits: [4:0]=i, [15:5]=s, [21:16]=bh.
// ---------------------------------------------------------------------------
__global__ __launch_bounds__(256) void rope_kernel(__bf16* __restrict__ Q,
                                                   __bf16* __restrict__ K,
                                                   const int* __restrict__ pos,
                                                   int perTensor) {
  int t = blockIdx.x * blockDim.x + threadIdx.x;
  __bf16* P = Q;
  if (t >= perTensor) { P = K; t -= perTensor; }
  const int i = t & 31;
  const int s = (t >> 5) & (S_LEN - 1);
  const size_t base = ((size_t)(t >> 16) * S_LEN + s) * D_K + 2 * i;
  const float ang = (float)pos[s] * __powf(10000.0f, -(float)i / 32.0f);
  float sn, cs;
  __sincosf(ang, &sn, &cs);
  const float x0 = bf2f(P[base]), x1 = bf2f(P[base + 1]);
  P[base]     = f2bf(cs * x0 - sn * x1);
  P[base + 1] = f2bf(sn * x0 + cs * x1);
}

// ---------------------------------------------------------------------------
// Flash attention, causal. One wave handles one (b,h, 16-row q tile).
// Q,K: [B,H,S,dk] bf16 (rope applied).  Vt: [B,H,dk,S] bf16.
// Output AO: [B,S,H*dk] bf16 (row-major [M,1024] for the final GEMM).
// 64-wide KV chunks. All 8 K-fragments (and 8 V-fragments) are hoisted into
// live arrays BEFORE the WMMA chains so the scheduler can clause the 16
// global_load_b128 together and overlap them with the WMMAs instead of the
// load -> s_wait_loadcnt 0 -> wmma lockstep seen when fragments are consumed
// as they are loaded.
// ---------------------------------------------------------------------------
__global__ __launch_bounds__(128) void attn_kernel(const __bf16* __restrict__ Q,
                                                   const __bf16* __restrict__ K,
                                                   const __bf16* __restrict__ Vt,
                                                   __bf16* __restrict__ AO) {
  __shared__ __bf16 Pl[4][16 * 64];  // per-wave P tile staging (2 KB each)

  const int lane = threadIdx.x & 31;
  const int w = threadIdx.x >> 5;
  const int l15 = lane & 15;
  const int hi = (lane >> 4) & 1;

  const int tIdx = blockIdx.x * 4 + w;      // 0 .. B*H*(S/16)-1
  const int qt = tIdx & 127;                // q tile within sequence
  const int bh = tIdx >> 7;                 // b*16 + h
  const int s0 = qt * 16;

  const __bf16* Qb = Q + (size_t)bh * S_LEN * D_K;
  const __bf16* Kb = K + (size_t)bh * S_LEN * D_K;
  const __bf16* Vb = Vt + (size_t)bh * D_K * S_LEN;

  // Q A-fragments (16x64 = two 16x32 chunks), loaded once
  const __bf16* qrow = Qb + (size_t)(s0 + l15) * D_K;
  const int kaOff = hi ? 16 : 0;
  ABf aq0, aq1;
  aq0.h[0] = *(const v8bf*)(qrow + kaOff);
  aq0.h[1] = *(const v8bf*)(qrow + kaOff + 8);
  aq1.h[0] = *(const v8bf*)(qrow + 32 + kaOff);
  aq1.h[1] = *(const v8bf*)(qrow + 32 + kaOff + 8);

  v8f acc[4] = {};
  float mrow[8], lrow[8];
#pragma unroll
  for (int r = 0; r < 8; ++r) { mrow[r] = -3.0e38f; lrow[r] = 0.0f; }

  const int tEnd = s0 + 16;
  for (int t0 = 0; t0 < tEnd; t0 += 64) {
    // ---- load ALL eight K-fragments for this 64-wide chunk up front ----
    ABf bk[4][2];
#pragma unroll
    for (int tt = 0; tt < 4; ++tt) {
      const __bf16* krow = Kb + (size_t)(t0 + tt * 16 + l15) * D_K + (hi ? 16 : 0);
      bk[tt][0].h[0] = *(const v8bf*)(krow);
      bk[tt][0].h[1] = *(const v8bf*)(krow + 8);
      bk[tt][1].h[0] = *(const v8bf*)(krow + 32);
      bk[tt][1].h[1] = *(const v8bf*)(krow + 40);
    }

    // ---- scores: four 16x16 tiles covering t0 .. t0+63 ----
    v8f sc[4];
#pragma unroll
    for (int tt = 0; tt < 4; ++tt) {
      v8f z = {};
      sc[tt] = WMMA_BF16(aq0.v, bk[tt][0].v, z);
      sc[tt] = WMMA_BF16(aq1.v, bk[tt][1].v, sc[tt]);
    }

    // ---- causal mask + online softmax over 64 columns ----
#pragma unroll
    for (int r = 0; r < 8; ++r) {
      const int srow = s0 + r + hi * 8;     // D layout: M = r + 8*(lane>>4)
      float x[4];
#pragma unroll
      for (int tt = 0; tt < 4; ++tt) {
        x[tt] = sc[tt][r] * 0.125f;         // 1/sqrt(64)
        if (t0 + tt * 16 + l15 > srow) x[tt] = -3.0e38f;
      }
      float mx = fmaxf(fmaxf(x[0], x[1]), fmaxf(x[2], x[3]));
      const float mnew = fmaxf(mrow[r], rowmax16(mx));
      const float corr = __expf(mrow[r] - mnew);
      mrow[r] = mnew;
      float e[4], esum = 0.0f;
#pragma unroll
      for (int tt = 0; tt < 4; ++tt) { e[tt] = __expf(x[tt] - mnew); esum += e[tt]; }
      lrow[r] = lrow[r] * corr + rowsum16(esum);
#pragma unroll
      for (int j = 0; j < 4; ++j) acc[j][r] *= corr;
      const int prow = r + hi * 8;
#pragma unroll
      for (int tt = 0; tt < 4; ++tt)
        Pl[w][prow * 64 + tt * 16 + l15] = f2bf(e[tt]);
    }

    // ---- load ALL eight V-fragments (contiguous thanks to V^T layout) ----
    ABf bv[4][2];
#pragma unroll
    for (int j = 0; j < 4; ++j) {
      const __bf16* vcol = Vb + (size_t)(j * 16 + l15) * S_LEN + t0 + (hi ? 16 : 0);
      bv[j][0].h[0] = *(const v8bf*)(vcol);
      bv[j][0].h[1] = *(const v8bf*)(vcol + 8);
      bv[j][1].h[0] = *(const v8bf*)(vcol + 32);
      bv[j][1].h[1] = *(const v8bf*)(vcol + 40);
    }

    // ---- re-layout P (D layout -> two 16x32 A-fragments) via LDS ----
    ABf ap0, ap1;
    ap0.h[0] = *(const v8bf*)&Pl[w][l15 * 64 + kaOff];
    ap0.h[1] = *(const v8bf*)&Pl[w][l15 * 64 + kaOff + 8];
    ap1.h[0] = *(const v8bf*)&Pl[w][l15 * 64 + 32 + kaOff];
    ap1.h[1] = *(const v8bf*)&Pl[w][l15 * 64 + 32 + kaOff + 8];

    // ---- acc += P @ V ----
#pragma unroll
    for (int j = 0; j < 4; ++j) {
      acc[j] = WMMA_BF16(ap0.v, bv[j][0].v, acc[j]);
      acc[j] = WMMA_BF16(ap1.v, bv[j][1].v, acc[j]);
    }
  }

  // ---- normalize and write [B,S,H*dk] bf16 ----
  const int b = bh >> 4, h = bh & 15;
#pragma unroll
  for (int r = 0; r < 8; ++r) {
    const int srow = s0 + r + hi * 8;
    const float inv = 1.0f / lrow[r];
#pragma unroll
    for (int j = 0; j < 4; ++j)
      AO[((size_t)b * S_LEN + srow) * D_MODEL + h * D_K + j * 16 + l15] =
          f2bf(acc[j][r] * inv);
  }
}

// ---------------------------------------------------------------------------
// Host-side launch
// ---------------------------------------------------------------------------
extern "C" void kernel_launch(void* const* d_in, const int* in_sizes, int n_in,
                              void* d_out, int out_size, void* d_ws, size_t ws_size,
                              hipStream_t stream) {
  (void)in_sizes; (void)n_in; (void)out_size; (void)ws_size;
  const float* x  = (const float*)d_in[0];
  const int* pos  = (const int*)d_in[1];
  const float* Wq = (const float*)d_in[2];
  const float* Wk = (const float*)d_in[3];
  const float* Wv = (const float*)d_in[4];
  const float* Wo = (const float*)d_in[5];

  const size_t nX = (size_t)M_ROWS * D_MODEL;          // 8,388,608
  const size_t nW = (size_t)D_MODEL * D_MODEL;         // 1,048,576

  char* ws = (char*)d_ws;
  __bf16* xb  = (__bf16*)ws;                 ws += nX * 2;   // x bf16 (reused as AO later)
  __bf16* Wqb = (__bf16*)ws;                 ws += nW * 2;
  __bf16* Wkb = (__bf16*)ws;                 ws += nW * 2;
  __bf16* Wvb = (__bf16*)ws;                 ws += nW * 2;
  __bf16* Wob = (__bf16*)ws;                 ws += nW * 2;
  __bf16* Qb  = (__bf16*)ws;                 ws += nX * 2;   // [B,H,S,dk]
  __bf16* Kb  = (__bf16*)ws;                 ws += nX * 2;   // [B,H,S,dk]
  __bf16* Vt  = (__bf16*)ws;                 ws += nX * 2;   // [B,H,dk,S]
  __bf16* AOb = xb;  // x bf16 dead after V projection; alias for attention output

  cvt_bf16<<<(int)(nX / (256 * 8)), 256, 0, stream>>>(x, xb);
  cvt_bf16<<<(int)(nW / (256 * 8)), 256, 0, stream>>>(Wq, Wqb);
  cvt_bf16<<<(int)(nW / (256 * 8)), 256, 0, stream>>>(Wk, Wkb);
  cvt_bf16<<<(int)(nW / (256 * 8)), 256, 0, stream>>>(Wv, Wvb);
  cvt_bf16<<<(int)(nW / (256 * 8)), 256, 0, stream>>>(Wo, Wob);

  dim3 gGemm(M_ROWS / 128, D_MODEL / 128);  // (64, 8)
  gemm_bf16<0><<<gGemm, 128, 0, stream>>>(xb, Wqb, (void*)Qb);
  gemm_bf16<0><<<gGemm, 128, 0, stream>>>(xb, Wkb, (void*)Kb);
  gemm_bf16<1><<<gGemm, 128, 0, stream>>>(xb, Wvb, (void*)Vt);

  const int pairsPerTensor = BATCH * N_HEADS * S_LEN * (D_K / 2);  // 4,194,304
  rope_kernel<<<(2 * pairsPerTensor) / 256, 256, 0, stream>>>(Qb, Kb, pos, pairsPerTensor);

  attn_kernel<<<(BATCH * N_HEADS * (S_LEN / 16)) / 4, 128, 0, stream>>>(Qb, Kb, Vt, AOb);

  gemm_bf16<2><<<gGemm, 128, 0, stream>>>(AOb, Wob, d_out);
}
